// GroupedSwiGLUExperts_86990267613558
// MI455X (gfx1250) — compile-verified, matrix operand order checked by reading, not run on hardware
//
#include <hip/hip_runtime.h>
#include <hip/hip_bf16.h>
#include <stddef.h>

// ---------------- problem constants ----------------
constexpr int MTOK   = 8192;   // tokens
constexpr int KDIM   = 2048;   // model dim (also down output dim)
constexpr int NDIM   = 1024;   // expert hidden dim
constexpr int NEXP   = 16;     // experts
constexpr int TOPK   = 2;
constexpr int TOTAL  = MTOK * TOPK;          // 16384 expanded rows
constexpr int TILE_M = 64;                   // rows per tile
constexpr int MAX_TILES = TOTAL / TILE_M + NEXP;   // 272 worst case
constexpr int MAX_TILES_PAD = 288;

typedef __attribute__((ext_vector_type(8)))  __bf16 v8bf;
typedef __attribute__((ext_vector_type(16))) __bf16 v16bf;
typedef __attribute__((ext_vector_type(8)))  float  v8f;

__device__ __forceinline__ unsigned short f32_to_bf16u(float f) {
    unsigned u = __builtin_bit_cast(unsigned, f);
    u += 0x7FFFu + ((u >> 16) & 1u);          // round-to-nearest-even
    return (unsigned short)(u >> 16);
}

#define SHUF16(lo, hi) __builtin_shufflevector(lo, hi, 0,1,2,3,4,5,6,7,8,9,10,11,12,13,14,15)

// =====================================================================
// Kernel 0: zero output (scatter-add target)
// =====================================================================
__global__ void zero_out_kernel(float* __restrict__ out, int n) {
    int i = blockIdx.x * blockDim.x + threadIdx.x;
    if (i < n) out[i] = 0.0f;
}

// =====================================================================
// Kernel 1: deterministic stable routing (single block, 256 threads)
// =====================================================================
__global__ void moe_route_kernel(const int* __restrict__ flat_idx,
                                 const float* __restrict__ flat_gate,
                                 int* __restrict__ offsets,      // [NEXP+1]
                                 int* __restrict__ numTiles,     // [1]
                                 int* __restrict__ tileExpert,   // [MAX_TILES_PAD]
                                 int* __restrict__ tileRowStart, // [MAX_TILES_PAD]
                                 int* __restrict__ rowToken,     // [TOTAL]
                                 float* __restrict__ rowGate)    // [TOTAL]
{
    __shared__ int cnt[NEXP];
    __shared__ int base[NEXP];
    __shared__ int chunkE[256];
    const int tid = threadIdx.x;

    if (tid < NEXP) cnt[tid] = 0;
    __syncthreads();
    for (int i = tid; i < TOTAL; i += 256)
        atomicAdd(&cnt[flat_idx[i]], 1);
    __syncthreads();

    if (tid == 0) {
        int acc = 0, t = 0;
        for (int e = 0; e < NEXP; ++e) {
            offsets[e] = acc;
            base[e]    = acc;
            const int c = cnt[e];
            for (int r = 0; r < c; r += TILE_M) {
                tileExpert[t]   = e;
                tileRowStart[t] = acc + r;
                ++t;
            }
            acc += c;
        }
        offsets[NEXP] = acc;
        *numTiles = t;
    }
    __syncthreads();

    // stable scatter in chunks of 256
    for (int c0 = 0; c0 < TOTAL; c0 += 256) {
        const int i = c0 + tid;
        const int e = flat_idx[i];
        chunkE[tid] = e;
        __syncthreads();
        int rank = 0;
        for (int j = 0; j < tid; ++j) rank += (chunkE[j] == e) ? 1 : 0;
        const int p = base[e] + rank;
        rowToken[p] = i / TOPK;
        rowGate[p]  = flat_gate[i];
        __syncthreads();
        if (tid < NEXP) {
            int add = 0;
            for (int j = 0; j < 256; ++j) add += (chunkE[j] == tid) ? 1 : 0;
            base[tid] += add;
        }
        __syncthreads();
    }
}

// =====================================================================
// Kernel 2: fused gate+up grouped GEMM + clamp/SiLU/mul -> hidden (bf16)
//   tile: 64 rows x 128 cols, block = 256 (8 waves)
//   wave: 4 M-subtiles x 1 N-subtile for BOTH gate and up (8 wmma/k-step)
//   double-buffered LDS + register prefetch, one barrier per k-step
// =====================================================================
__global__ void __launch_bounds__(256)
gemm_gateup_kernel(const float* __restrict__ flat_h,    // [MTOK,KDIM]
                   const float* __restrict__ gate_w,    // [NEXP,NDIM,KDIM]
                   const float* __restrict__ up_w,      // [NEXP,NDIM,KDIM]
                   const int* __restrict__ offsets,
                   const int* __restrict__ numTiles,
                   const int* __restrict__ tileExpert,
                   const int* __restrict__ tileRowStart,
                   const int* __restrict__ rowToken,
                   unsigned short* __restrict__ hidden) // [TOTAL,NDIM] bf16
{
    __shared__ __align__(16) unsigned short ldsA [2][TILE_M * 32];  //  8 KB
    __shared__ __align__(16) unsigned short ldsBg[2][128 * 32];     // 16 KB
    __shared__ __align__(16) unsigned short ldsBu[2][128 * 32];     // 16 KB

    const int t = blockIdx.x;
    if (t >= *numTiles) return;

    const int e        = tileExpert[t];
    const int rowStart = tileRowStart[t];
    const int tileRows = min(TILE_M, offsets[e + 1] - rowStart);
    const int nBase    = blockIdx.y * 128;

    const int tid  = threadIdx.x;
    const int lane = tid & 31;
    const int wid  = tid >> 5;

    const float* wg = gate_w + (size_t)e * NDIM * KDIM;
    const float* wu = up_w   + (size_t)e * NDIM * KDIM;

    // ---- loop-invariant addressing: A row pointers (gathered), B offsets
    const float* aPtr[8];       // nullptr => padded row (write zero)
    #pragma unroll
    for (int j = 0; j < 8; ++j) {
        const int idx = j * 256 + tid;
        const int r   = idx >> 5;
        const int kk  = idx & 31;
        aPtr[j] = (r < tileRows)
                ? flat_h + (size_t)rowToken[rowStart + r] * KDIM + kk
                : nullptr;
    }
    size_t bOff[16];
    #pragma unroll
    for (int j = 0; j < 16; ++j) {
        const int idx = j * 256 + tid;
        bOff[j] = (size_t)(nBase + (idx >> 5)) * KDIM + (idx & 31);
    }

    float aR[8], bgR[16], buR[16];
    auto loadRegs = [&](int k0) {
        #pragma unroll
        for (int j = 0; j < 8; ++j)
            aR[j] = aPtr[j] ? aPtr[j][k0] : 0.0f;
        #pragma unroll
        for (int j = 0; j < 16; ++j) {
            bgR[j] = wg[bOff[j] + k0];
            buR[j] = wu[bOff[j] + k0];
        }
    };
    auto storeRegs = [&](int buf) {
        #pragma unroll
        for (int j = 0; j < 8; ++j)
            ldsA[buf][j * 256 + tid] = f32_to_bf16u(aR[j]);
        #pragma unroll
        for (int j = 0; j < 16; ++j) {
            ldsBg[buf][j * 256 + tid] = f32_to_bf16u(bgR[j]);
            ldsBu[buf][j * 256 + tid] = f32_to_bf16u(buR[j]);
        }
    };

    v8f cg[4] = {};
    v8f cu[4] = {};

    constexpr int KSTEPS = KDIM / 32;       // 64
    loadRegs(0);
    storeRegs(0);

    const int arow = lane & 15;
    const int kgrp = (lane >> 4) * 8;

    for (int ks = 0; ks < KSTEPS; ++ks) {
        const int  cur      = ks & 1;
        const bool haveNext = (ks + 1) < KSTEPS;
        if (haveNext) loadRegs((ks + 1) * 32);   // prefetch (no LDS touch)

        __syncthreads();                          // buf[cur] ready; buf[cur^1] free

        v16bf a[4];
        #pragma unroll
        for (int s = 0; s < 4; ++s) {
            const unsigned short* p = &ldsA[cur][(s * 16 + arow) * 32];
            const v8bf lo = *reinterpret_cast<const v8bf*>(p + kgrp);
            const v8bf hi = *reinterpret_cast<const v8bf*>(p + 16 + kgrp);
            a[s] = SHUF16(lo, hi);
        }
        const unsigned short* pg = &ldsBg[cur][(wid * 16 + arow) * 32];
        const v16bf bg = SHUF16(*reinterpret_cast<const v8bf*>(pg + kgrp),
                                *reinterpret_cast<const v8bf*>(pg + 16 + kgrp));
        const unsigned short* pu = &ldsBu[cur][(wid * 16 + arow) * 32];
        const v16bf bu = SHUF16(*reinterpret_cast<const v8bf*>(pu + kgrp),
                                *reinterpret_cast<const v8bf*>(pu + 16 + kgrp));

        #pragma unroll
        for (int s = 0; s < 4; ++s) {
            cg[s] = __builtin_amdgcn_wmma_f32_16x16x32_bf16(false, a[s], false, bg,
                                                            (short)0, cg[s], false, false);
            cu[s] = __builtin_amdgcn_wmma_f32_16x16x32_bf16(false, a[s], false, bu,
                                                            (short)0, cu[s], false, false);
        }

        if (haveNext) storeRegs(cur ^ 1);
    }

    // ---- epilogue: clamp, SiLU, multiply, store bf16 hidden
    const int nOut  = nBase + wid * 16 + (lane & 15);
    const int mBase = (lane >> 4) * 8;
    #pragma unroll
    for (int s = 0; s < 4; ++s) {
        #pragma unroll
        for (int v = 0; v < 8; ++v) {
            const int m = s * 16 + mBase + v;
            if (m < tileRows) {
                float g = fminf(cg[s][v], 10.0f);
                float u = fminf(fmaxf(cu[s][v], -10.0f), 10.0f);
                float h = (g / (1.0f + __expf(-g))) * u;   // silu(g)*u
                hidden[(size_t)(rowStart + m) * NDIM + nOut] = f32_to_bf16u(h);
            }
        }
    }
}

// =====================================================================
// Kernel 3: down grouped GEMM, scale by gate, scatter-add into out
//   tile: 64 rows x 128 ko, block = 256 (8 waves), 4 wmma/wave/step
// =====================================================================
__global__ void __launch_bounds__(256)
gemm_down_kernel(const unsigned short* __restrict__ hidden, // [TOTAL,NDIM] bf16
                 const float* __restrict__ down_w,           // [NEXP,KDIM,NDIM]
                 const int* __restrict__ offsets,
                 const int* __restrict__ numTiles,
                 const int* __restrict__ tileExpert,
                 const int* __restrict__ tileRowStart,
                 const int* __restrict__ rowToken,
                 const float* __restrict__ rowGate,
                 float* __restrict__ out)                    // [MTOK,KDIM]
{
    __shared__ __align__(16) unsigned short ldsA[2][TILE_M * 32];  //  8 KB
    __shared__ __align__(16) unsigned short ldsB[2][128 * 32];     // 16 KB

    const int t = blockIdx.x;
    if (t >= *numTiles) return;

    const int e        = tileExpert[t];
    const int rowStart = tileRowStart[t];
    const int tileRows = min(TILE_M, offsets[e + 1] - rowStart);
    const int koBase   = blockIdx.y * 128;

    const int tid  = threadIdx.x;
    const int lane = tid & 31;
    const int wid  = tid >> 5;

    const float* wd = down_w + (size_t)e * KDIM * NDIM;

    // loop-invariant addressing
    const unsigned short* aPtr[8];
    #pragma unroll
    for (int j = 0; j < 8; ++j) {
        const int idx = j * 256 + tid;
        const int r   = idx >> 5;
        aPtr[j] = (r < tileRows)
                ? hidden + (size_t)(rowStart + r) * NDIM + (idx & 31)
                : nullptr;
    }
    size_t bOff[16];
    #pragma unroll
    for (int j = 0; j < 16; ++j) {
        const int idx = j * 256 + tid;
        bOff[j] = (size_t)(koBase + (idx >> 5)) * NDIM + (idx & 31);
    }

    unsigned short aR[8];
    float bR[16];
    auto loadRegs = [&](int n0) {
        #pragma unroll
        for (int j = 0; j < 8; ++j)
            aR[j] = aPtr[j] ? aPtr[j][n0] : (unsigned short)0;
        #pragma unroll
        for (int j = 0; j < 16; ++j)
            bR[j] = wd[bOff[j] + n0];
    };
    auto storeRegs = [&](int buf) {
        #pragma unroll
        for (int j = 0; j < 8; ++j)
            ldsA[buf][j * 256 + tid] = aR[j];
        #pragma unroll
        for (int j = 0; j < 16; ++j)
            ldsB[buf][j * 256 + tid] = f32_to_bf16u(bR[j]);
    };

    v8f acc[4] = {};

    constexpr int NSTEPS = NDIM / 32;       // 32
    loadRegs(0);
    storeRegs(0);

    const int arow = lane & 15;
    const int kgrp = (lane >> 4) * 8;

    for (int ks = 0; ks < NSTEPS; ++ks) {
        const int  cur      = ks & 1;
        const bool haveNext = (ks + 1) < NSTEPS;
        if (haveNext) loadRegs((ks + 1) * 32);

        __syncthreads();

        v16bf a[4];
        #pragma unroll
        for (int s = 0; s < 4; ++s) {
            const unsigned short* p = &ldsA[cur][(s * 16 + arow) * 32];
            a[s] = SHUF16(*reinterpret_cast<const v8bf*>(p + kgrp),
                          *reinterpret_cast<const v8bf*>(p + 16 + kgrp));
        }
        const unsigned short* pb = &ldsB[cur][(wid * 16 + arow) * 32];
        const v16bf b = SHUF16(*reinterpret_cast<const v8bf*>(pb + kgrp),
                               *reinterpret_cast<const v8bf*>(pb + 16 + kgrp));

        #pragma unroll
        for (int s = 0; s < 4; ++s)
            acc[s] = __builtin_amdgcn_wmma_f32_16x16x32_bf16(false, a[s], false, b,
                                                             (short)0, acc[s], false, false);

        if (haveNext) storeRegs(cur ^ 1);
    }

    // ---- epilogue: scale by gate, scatter-add per token
    const int ko    = koBase + wid * 16 + (lane & 15);
    const int mBase = (lane >> 4) * 8;
    #pragma unroll
    for (int s = 0; s < 4; ++s) {
        #pragma unroll
        for (int v = 0; v < 8; ++v) {
            const int m = s * 16 + mBase + v;
            if (m < tileRows) {
                const float g  = rowGate[rowStart + m];
                const int  tok = rowToken[rowStart + m];
                atomicAdd(&out[(size_t)tok * KDIM + ko], acc[s][v] * g);
            }
        }
    }
}

// =====================================================================
// launch
// =====================================================================
extern "C" void kernel_launch(void* const* d_in, const int* in_sizes, int n_in,
                              void* d_out, int out_size, void* d_ws, size_t ws_size,
                              hipStream_t stream) {
    const float* flat_h    = (const float*)d_in[0];
    const int*   flat_idx  = (const int*)  d_in[1];
    const float* flat_gate = (const float*)d_in[2];
    const float* gate_w    = (const float*)d_in[3];
    const float* up_w      = (const float*)d_in[4];
    const float* down_w    = (const float*)d_in[5];
    float* out = (float*)d_out;

    // workspace layout
    int*   offsets      = (int*)d_ws;                    // 17 (pad to 32)
    int*   numTiles     = offsets + 17;
    int*   tileExpert   = offsets + 32;                  // MAX_TILES_PAD
    int*   tileRowStart = tileExpert + MAX_TILES_PAD;
    int*   rowToken     = tileRowStart + MAX_TILES_PAD;  // TOTAL
    float* rowGate      = (float*)(rowToken + TOTAL);    // TOTAL
    unsigned short* hidden = (unsigned short*)(rowGate + TOTAL); // TOTAL*NDIM bf16

    const int outElems = MTOK * KDIM;
    zero_out_kernel<<<(outElems + 1023) / 1024, 1024, 0, stream>>>(out, outElems);

    moe_route_kernel<<<1, 256, 0, stream>>>(flat_idx, flat_gate, offsets, numTiles,
                                            tileExpert, tileRowStart, rowToken, rowGate);

    gemm_gateup_kernel<<<dim3(MAX_TILES, NDIM / 128), 256, 0, stream>>>(
        flat_h, gate_w, up_w, offsets, numTiles, tileExpert, tileRowStart,
        rowToken, hidden);

    gemm_down_kernel<<<dim3(MAX_TILES, KDIM / 128), 256, 0, stream>>>(
        hidden, down_w, offsets, numTiles, tileExpert, tileRowStart,
        rowToken, rowGate, out);
}
